// MultiHeadAttention_24137716203657
// MI455X (gfx1250) — matrix operand order repaired and checked by reading, not compiled, not run on hardware
//
#include <hip/hip_runtime.h>

// ---------------- types & helpers ----------------
typedef __attribute__((ext_vector_type(16))) __bf16 v16bf;
typedef __attribute__((ext_vector_type(8)))  __bf16 v8bf;
typedef __attribute__((ext_vector_type(4)))  __bf16 v4bf;
typedef __attribute__((ext_vector_type(8)))  float  v8f;
typedef __attribute__((ext_vector_type(4)))  int    v4i;
typedef __attribute__((address_space(1))) v4i* as1_v4i;
typedef __attribute__((address_space(3))) v4i* as3_v4i;

#define CAT16(lo, hi) __builtin_shufflevector((lo), (hi), 0,1,2,3,4,5,6,7,8,9,10,11,12,13,14,15)

static __device__ __forceinline__ v8f wmma_bf16(v16bf a, v16bf b, v8f c) {
  return __builtin_amdgcn_wmma_f32_16x16x32_bf16(false, a, false, b, (short)0, c, false, false);
}

// gfx1250 async global->LDS path (guarded; sync ds_store fallback keeps compile safe)
#if defined(__has_builtin)
# if __has_builtin(__builtin_amdgcn_global_load_async_to_lds_b128)
#  define HAS_ASYNC_LDS 1
# endif
#endif

static __device__ __forceinline__ void async_wait0() {
#if defined(HAS_ASYNC_LDS)
# if __has_builtin(__builtin_amdgcn_s_wait_asynccnt)
  __builtin_amdgcn_s_wait_asynccnt(0);
# else
  asm volatile("s_wait_asynccnt 0x0" ::: "memory");
# endif
#endif
}

static __device__ __forceinline__ void copy_b128_to_lds(const __bf16* gsrc, __bf16* ldst) {
#if defined(HAS_ASYNC_LDS)
  __builtin_amdgcn_global_load_async_to_lds_b128((as1_v4i)gsrc, (as3_v4i)ldst, 0, 0);
#else
  *(v8bf*)ldst = *(const v8bf*)gsrc;
#endif
}

// dims
#define DE 1024
#define TT 2048
#define BB 4
#define NH 16
#define DH 64
#define MM (BB * TT)   // 8192 rows

// ---------------- kernel 1: fp32 -> bf16 convert (vectorized x4) ----------------
__global__ void cvt_f32_bf16(const float* __restrict__ src, __bf16* __restrict__ dst, int n4) {
  int idx = blockIdx.x * blockDim.x + threadIdx.x;
  int stride = gridDim.x * blockDim.x;
  for (int i = idx; i < n4; i += stride) {
    float4 f = ((const float4*)src)[i];
    v4bf o;
    o[0] = (__bf16)f.x; o[1] = (__bf16)f.y; o[2] = (__bf16)f.z; o[3] = (__bf16)f.w;
    ((v4bf*)dst)[i] = o;
  }
}

// ---------------- kernel 2: QKV projections ----------------
// grid (64, 16, 3), block 256 (8 waves). Wave computes 16x64 tile of x @ W^T.
// Block cooperatively stages the shared 64x32 B tile in LDS (double buffered, async).
// z=0 -> Q [B,H,T,DH]; z=1 -> K [B,H,T,DH]; z=2 -> V transposed [B,H,DH,T]
__global__ __launch_bounds__(256) void qkv_proj(const __bf16* __restrict__ xb,
                                                const __bf16* __restrict__ Wq,
                                                const __bf16* __restrict__ Wk,
                                                const __bf16* __restrict__ Wv,
                                                __bf16* __restrict__ Qb,
                                                __bf16* __restrict__ Kb,
                                                __bf16* __restrict__ Vtb) {
  __shared__ __attribute__((aligned(32))) __bf16 ldsB[2][64][32];
  const int tid  = threadIdx.x;
  const int lane = tid & 31;
  const int wave = tid >> 5;
  const int col  = lane & 15;
  const int g    = lane >> 4;
  const int mBase = (blockIdx.x * 8 + wave) * 16;   // 0..8191
  const int nBase = blockIdx.y * 64;                // 0..960
  const int z = blockIdx.z;
  const __bf16* W = (z == 0) ? Wq : ((z == 1) ? Wk : Wv);

  const __bf16* arow = xb + (size_t)(mBase + col) * DE;
  // cooperative B staging: thread -> (row, 16B chunk)
  const int brow   = tid >> 2;          // 0..63
  const int bchunk = (tid & 3) * 8;     // 0,8,16,24
  const __bf16* bsrc = W + (size_t)(nBase + brow) * DE + bchunk;

  // prefetch k=0 B tile + first A fragment
  copy_b128_to_lds(bsrc, &ldsB[0][brow][bchunk]);
  v16bf afCur;
  {
    v8bf lo = *(const v8bf*)(arow + 16 * g);
    v8bf hi = *(const v8bf*)(arow + 8 + 16 * g);
    afCur = CAT16(lo, hi);
  }

  v8f acc[4] = {};
  for (int k = 0; k < DE; k += 32) {
    const int cur = (k >> 5) & 1;
    async_wait0();
    __syncthreads();                      // buf[cur] ready; prior reads of buf[cur^1] done
    if (k + 32 < DE)
      copy_b128_to_lds(bsrc + k + 32, &ldsB[cur ^ 1][brow][bchunk]);
    v16bf afNext = afCur;
    if (k + 32 < DE) {
      v8bf lo = *(const v8bf*)(arow + k + 32 + 16 * g);
      v8bf hi = *(const v8bf*)(arow + k + 40 + 16 * g);
      afNext = CAT16(lo, hi);
    }
#pragma unroll
    for (int nt = 0; nt < 4; ++nt) {
      v16bf bf = *(const v16bf*)&ldsB[cur][nt * 16 + col][16 * g];
      acc[nt] = wmma_bf16(afCur, bf, acc[nt]);
    }
    afCur = afNext;
  }

#pragma unroll
  for (int nt = 0; nt < 4; ++nt) {
#pragma unroll
    for (int i = 0; i < 8; ++i) {
      const int m = mBase + i + 8 * g;           // global row (b*T + t)
      const int n = nBase + nt * 16 + col;       // global col (h*DH + dh)
      const int b = m >> 11, t = m & (TT - 1);
      const int h = n >> 6, dh = n & (DH - 1);
      const __bf16 val = (__bf16)acc[nt][i];
      if (z == 0)      Qb [(((size_t)b * NH + h) * TT + t) * DH + dh] = val;
      else if (z == 1) Kb [(((size_t)b * NH + h) * TT + t) * DH + dh] = val;
      else             Vtb[(((size_t)b * NH + h) * DH + dh) * TT + t] = val;
    }
  }
}

// ---------------- kernel 3: causal flash attention ----------------
// grid (16, 64), block 256. Wave w handles query tile (blockIdx.x*8+w) of head blockIdx.y.
__global__ __launch_bounds__(256) void attn_flash(const __bf16* __restrict__ Qb,
                                                  const __bf16* __restrict__ Kb,
                                                  const __bf16* __restrict__ Vtb,
                                                  __bf16* __restrict__ ctxb) {
  __shared__ __attribute__((aligned(32))) __bf16 lds[8 * 16 * 32];
  const int lane = threadIdx.x & 31;
  const int wave = threadIdx.x >> 5;
  const int col  = lane & 15;
  const int g    = lane >> 4;
  __bf16* stage = &lds[wave * 16 * 32];

  const int qtile = blockIdx.x * 8 + wave;   // 0..127
  const int bh    = blockIdx.y;              // 0..63
  const int b = bh >> 4, h = bh & 15;
  const __bf16* Qp = Qb  + (size_t)bh * TT * DH;
  const __bf16* Kp = Kb  + (size_t)bh * TT * DH;
  const __bf16* Vp = Vtb + (size_t)bh * DH * TT;
  const int qBase = qtile * 16;

  // Q fragments (A layout): two 16x32 chunks over dh
  v16bf qf[2];
  {
    const __bf16* qrow = Qp + (size_t)(qBase + col) * DH;
#pragma unroll
    for (int f = 0; f < 2; ++f) {
      v8bf lo = *(const v8bf*)(qrow + 32 * f + 16 * g);
      v8bf hi = *(const v8bf*)(qrow + 32 * f + 8 + 16 * g);
      qf[f] = CAT16(lo, hi);
    }
  }

  v8f acc0 = {}, acc1 = {}, acc2 = {}, acc3 = {};
  float mrow[8], lrow[8];
#pragma unroll
  for (int i = 0; i < 8; ++i) { mrow[i] = -1e30f; lrow[i] = 0.0f; }
  const float LOG2E = 1.44269504089f;
  const float SCALE = 0.125f;  // 1/sqrt(DH)

  for (int j = 0; j < qBase + 16; j += 32) {
    // ---- issue ALL K and V fragment loads up front (V overlaps softmax below)
    v16bf kb[2][2], vf[4];
#pragma unroll
    for (int f = 0; f < 2; ++f) {
      kb[0][f] = *(const v16bf*)(Kp + (size_t)(j + col) * DH + 32 * f + 16 * g);
      kb[1][f] = *(const v16bf*)(Kp + (size_t)(j + 16 + col) * DH + 32 * f + 16 * g);
    }
#pragma unroll
    for (int nt = 0; nt < 4; ++nt)
      vf[nt] = *(const v16bf*)(Vp + (size_t)(nt * 16 + col) * TT + j + 16 * g);

    // ---- scores: S0 (keys j..j+15), S1 (keys j+16..j+31)
    v8f s0 = {}, s1 = {};
#pragma unroll
    for (int f = 0; f < 2; ++f) {
      s0 = wmma_bf16(qf[f], kb[0][f], s0);
      s1 = wmma_bf16(qf[f], kb[1][f], s1);
    }

    // ---- online softmax (per-row stats; row spans one 16-lane group)
    float alpha[8];
#pragma unroll
    for (int i = 0; i < 8; ++i) {
      const int row = qBase + i + 8 * g;
      float x0 = (j + col      <= row) ? s0[i] * SCALE : -1e30f;
      float x1 = (j + 16 + col <= row) ? s1[i] * SCALE : -1e30f;
      float rmax = fmaxf(x0, x1);
#pragma unroll
      for (int mk = 1; mk < 16; mk <<= 1) rmax = fmaxf(rmax, __shfl_xor(rmax, mk, 32));
      const float mnew = fmaxf(mrow[i], rmax);
      const float a  = __builtin_exp2f((mrow[i] - mnew) * LOG2E);
      const float e0 = __builtin_exp2f((x0 - mnew) * LOG2E);
      const float e1 = __builtin_exp2f((x1 - mnew) * LOG2E);
      float rs = e0 + e1;
#pragma unroll
      for (int mk = 1; mk < 16; mk <<= 1) rs += __shfl_xor(rs, mk, 32);
      lrow[i] = lrow[i] * a + rs;
      mrow[i] = mnew;
      alpha[i] = a;
      // stage P (C layout -> row-major 16x32 tile in LDS)
      const int prow = i + 8 * g;
      stage[prow * 32 + col]      = (__bf16)e0;
      stage[prow * 32 + 16 + col] = (__bf16)e1;
    }
#pragma unroll
    for (int i = 0; i < 8; ++i) {
      acc0[i] *= alpha[i]; acc1[i] *= alpha[i];
      acc2[i] *= alpha[i]; acc3[i] *= alpha[i];
    }

    // ---- reload P in A-fragment layout (same wave; DS ordering guarantees)
    v16bf pf;
    {
      const __bf16* prow = stage + col * 32;
      v8bf lo = *(const v8bf*)(prow + 16 * g);
      v8bf hi = *(const v8bf*)(prow + 8 + 16 * g);
      pf = CAT16(lo, hi);
    }

    // ---- ctx += P @ V (V fragments already in flight since loop top)
    acc0 = wmma_bf16(pf, vf[0], acc0);
    acc1 = wmma_bf16(pf, vf[1], acc1);
    acc2 = wmma_bf16(pf, vf[2], acc2);
    acc3 = wmma_bf16(pf, vf[3], acc3);
  }

  // ---- epilogue: normalize and write ctx in [B,T, h*64+dh] bf16 (GEMM-ready)
#pragma unroll
  for (int i = 0; i < 8; ++i) {
    const float inv = 1.0f / lrow[i];
    const int row = qBase + i + 8 * g;
    const size_t base = ((size_t)b * TT + row) * DE + (size_t)h * DH;
    ctxb[base +  0 + col] = (__bf16)(acc0[i] * inv);
    ctxb[base + 16 + col] = (__bf16)(acc1[i] * inv);
    ctxb[base + 32 + col] = (__bf16)(acc2[i] * inv);
    ctxb[base + 48 + col] = (__bf16)(acc3[i] * inv);
  }
}

// ---------------- kernel 4: output projection + bias (fp32 out) ----------------
// grid (64, 16), block 256. Same LDS-staged, double-buffered GEMM as qkv_proj.
__global__ __launch_bounds__(256) void o_proj(const __bf16* __restrict__ ctxb,
                                              const __bf16* __restrict__ Wo,
                                              const float* __restrict__ bo,
                                              float* __restrict__ out) {
  __shared__ __attribute__((aligned(32))) __bf16 ldsB[2][64][32];
  const int tid  = threadIdx.x;
  const int lane = tid & 31;
  const int wave = tid >> 5;
  const int col  = lane & 15;
  const int g    = lane >> 4;
  const int mBase = (blockIdx.x * 8 + wave) * 16;
  const int nBase = blockIdx.y * 64;

  const __bf16* arow = ctxb + (size_t)(mBase + col) * DE;
  const int brow   = tid >> 2;
  const int bchunk = (tid & 3) * 8;
  const __bf16* bsrc = Wo + (size_t)(nBase + brow) * DE + bchunk;

  copy_b128_to_lds(bsrc, &ldsB[0][brow][bchunk]);
  v16bf afCur;
  {
    v8bf lo = *(const v8bf*)(arow + 16 * g);
    v8bf hi = *(const v8bf*)(arow + 8 + 16 * g);
    afCur = CAT16(lo, hi);
  }

  v8f acc[4] = {};
  for (int k = 0; k < DE; k += 32) {
    const int cur = (k >> 5) & 1;
    async_wait0();
    __syncthreads();
    if (k + 32 < DE)
      copy_b128_to_lds(bsrc + k + 32, &ldsB[cur ^ 1][brow][bchunk]);
    v16bf afNext = afCur;
    if (k + 32 < DE) {
      v8bf lo = *(const v8bf*)(arow + k + 32 + 16 * g);
      v8bf hi = *(const v8bf*)(arow + k + 40 + 16 * g);
      afNext = CAT16(lo, hi);
    }
#pragma unroll
    for (int nt = 0; nt < 4; ++nt) {
      v16bf bf = *(const v16bf*)&ldsB[cur][nt * 16 + col][16 * g];
      acc[nt] = wmma_bf16(afCur, bf, acc[nt]);
    }
    afCur = afNext;
  }

#pragma unroll
  for (int nt = 0; nt < 4; ++nt) {
#pragma unroll
    for (int i = 0; i < 8; ++i) {
      const int m = mBase + i + 8 * g;
      const int n = nBase + nt * 16 + col;
      out[(size_t)m * DE + n] = acc[nt][i] + bo[n];
    }
  }
}

// ---------------- host launch ----------------
extern "C" void kernel_launch(void* const* d_in, const int* in_sizes, int n_in,
                              void* d_out, int out_size, void* d_ws, size_t ws_size,
                              hipStream_t stream) {
  const float* x  = (const float*)d_in[0];
  const float* Wq = (const float*)d_in[1];
  const float* Wk = (const float*)d_in[2];
  const float* Wv = (const float*)d_in[3];
  const float* Wo = (const float*)d_in[4];
  const float* bo = (const float*)d_in[5];
  float* out = (float*)d_out;

  char* w = (char*)d_ws;
  __bf16* xb   = (__bf16*)w; w += (size_t)MM * DE * 2;        // 16 MiB
  __bf16* Wqb  = (__bf16*)w; w += (size_t)DE * DE * 2;        //  2 MiB
  __bf16* Wkb  = (__bf16*)w; w += (size_t)DE * DE * 2;
  __bf16* Wvb  = (__bf16*)w; w += (size_t)DE * DE * 2;
  __bf16* Wob  = (__bf16*)w; w += (size_t)DE * DE * 2;
  __bf16* Qb   = (__bf16*)w; w += (size_t)MM * DE * 2;        // [B,H,T,DH]
  __bf16* Kb   = (__bf16*)w; w += (size_t)MM * DE * 2;        // [B,H,T,DH]
  __bf16* Vtb  = (__bf16*)w; w += (size_t)MM * DE * 2;        // [B,H,DH,T]
  __bf16* ctxb = (__bf16*)w; w += (size_t)MM * DE * 2;        // [B*T, DE]

  // 1) converts (x4 vectorized)
  cvt_f32_bf16<<<1024, 256, 0, stream>>>(x,  xb,  MM * DE / 4);
  cvt_f32_bf16<<<256,  256, 0, stream>>>(Wq, Wqb, DE * DE / 4);
  cvt_f32_bf16<<<256,  256, 0, stream>>>(Wk, Wkb, DE * DE / 4);
  cvt_f32_bf16<<<256,  256, 0, stream>>>(Wv, Wvb, DE * DE / 4);
  cvt_f32_bf16<<<256,  256, 0, stream>>>(Wo, Wob, DE * DE / 4);

  // 2) Q/K/V projections (z selects matrix)
  qkv_proj<<<dim3(MM / 16 / 8, DE / 64, 3), 256, 0, stream>>>(xb, Wqb, Wkb, Wvb, Qb, Kb, Vtb);

  // 3) causal flash attention
  attn_flash<<<dim3(TT / 16 / 8, BB * NH), 256, 0, stream>>>(Qb, Kb, Vtb, ctxb);

  // 4) output projection + bias
  o_proj<<<dim3(MM / 16 / 8, DE / 64), 256, 0, stream>>>(ctxb, Wob, bo, out);
}
